// Classifier_6528350290016
// MI455X (gfx1250) — matrix operand order repaired and checked by reading, not compiled, not run on hardware
//
#include <hip/hip_runtime.h>
#include <stdint.h>

// ---------------------------------------------------------------------------
// Edge-MLP link predictor as a bf16 WMMA GEMM.
//   M = E (edges, tiled 256/block), K = 256 (concat feat), N = 128 (hidden)
//   Each of 8 waves owns TWO 16-row M blocks -> every B fragment read from
//   LDS feeds two independent v_wmma ops (halves LDS traffic per WMMA).
//   Epilogue (bias, ReLU, dot W2, butterfly reduce) fused in registers.
// ---------------------------------------------------------------------------

typedef __attribute__((ext_vector_type(8)))  unsigned short v8us;
typedef __attribute__((ext_vector_type(16))) unsigned short v16us;
typedef __attribute__((ext_vector_type(16))) __bf16        v16bf;
typedef __attribute__((ext_vector_type(8)))  float         v8f;

#define HDIM   128          // hidden width / N
#define KDIM   256          // 2*H concat width / K
#define TILE_E 256          // edges per block / M tile (16 blocks of 16)
#define LDA    264          // padded A row stride in bf16 elems (bank-conflict pad)
#define SMEM_BYTES ((TILE_E * LDA + KDIM * HDIM) * 2 + 2 * HDIM * 4)

static __device__ __forceinline__ unsigned short f32_to_bf16(float f) {
    unsigned int u = __builtin_bit_cast(unsigned int, f);
    u += 0x7FFFu + ((u >> 16) & 1u);          // round-to-nearest-even
    return (unsigned short)(u >> 16);
}

static __device__ __forceinline__ v16bf ldsA_frag(const unsigned short* p) {
    const v8us lo = *(const v8us*)p;          // K = kb+0..7   (this lane half)
    const v8us hi = *(const v8us*)(p + 16);   // K = kb+16..23 (this lane half)
    const v16us u = __builtin_shufflevector(lo, hi,
                        0,1,2,3,4,5,6,7,8,9,10,11,12,13,14,15);
    return __builtin_bit_cast(v16bf, u);
}

__global__ void __launch_bounds__(256)
edge_mlp_wmma_kernel(const float* __restrict__ x_nc,
                     const float* __restrict__ x_pr,
                     const long long* __restrict__ eidx,   // [2, E] int64
                     const float* __restrict__ W1,         // [256,128] row-major
                     const float* __restrict__ b1,         // [128]
                     const float* __restrict__ W2,         // [128]
                     const float* __restrict__ b2,         // [1]
                     float* __restrict__ out,              // [E]
                     int E) {
    extern __shared__ unsigned short smem[];
    unsigned short* ldsA  = smem;                          // [TILE_E][LDA] bf16
    unsigned short* ldsB  = smem + TILE_E * LDA;           // B fragments, 32768 bf16
    float*          ldsW2 = (float*)(smem + TILE_E * LDA + KDIM * HDIM);
    float*          ldsB1 = ldsW2 + HDIM;

    const int tid  = threadIdx.x;
    const int wv   = tid >> 5;                 // wave id (0..7)
    const int lane = tid & 31;
    const long long ebase = (long long)blockIdx.x * TILE_E;

    // ---- Phase 0a: small vectors ------------------------------------------
    if (tid < HDIM) { ldsW2[tid] = W2[tid]; ldsB1[tid] = b1[tid]; }

    // ---- Phase 0b: W1 -> bf16, pre-swizzled into B-fragment layout --------
    // Fragment (nb,kc): lane l<16 holds N=nb*16+l,      K=kc*32+0..7,16..23;
    //                   lane l>=16 holds N=nb*16+(l-16), K=kc*32+8..15,24..31.
    for (int g = tid; g < KDIM * HDIM; g += 256) {
        const int k  = g >> 7;
        const int n  = g & (HDIM - 1);
        const int nb = n >> 4;
        const int kc = k >> 5;
        const int fl = (n & 15) | (((k >> 3) & 1) << 4);
        ldsB[(((((nb << 3) | kc) << 5) | fl) << 4) | ((((k >> 4) & 1) << 3) | (k & 7))] = f32_to_bf16(W1[g]);
    }

    // ---- Phase 1: gather + convert edge features (coalesced per row) ------
    // 512 rows: row r -> edge r>>1, half r&1 (0 = ncRNA cols 0..127, 1 = protein).
    for (int it = 0; it < 64; ++it) {
        const int r  = (wv << 6) | it;
        const int el = r >> 1;
        const int hf = r & 1;
        long long e  = ebase + el;
        long long ec = (e < (long long)E) ? e : 0;       // clamp tail
        const long long idx = hf ? eidx[(long long)E + ec] : eidx[ec];
        const float4* src = (const float4*)((hf ? x_pr : x_nc) + idx * HDIM);
        const float4 v = src[lane];                      // 512B coalesced row load
        ushort4 p;
        p.x = f32_to_bf16(v.x); p.y = f32_to_bf16(v.y);
        p.z = f32_to_bf16(v.z); p.w = f32_to_bf16(v.w);
        *(ushort4*)&ldsA[el * LDA + (hf << 7) + (lane << 2)] = p;
    }
    __syncthreads();

    // ---- Phase 2: 128 x v_wmma_f32_16x16x32_bf16 per wave -----------------
    // Wave wv computes M blocks wv and wv+8; one B fragment -> two WMMAs.
    v8f acc0[8] = {};
    v8f acc1[8] = {};
    const int l15  = lane & 15;
    const int hsel = lane >> 4;
    const unsigned short* arow0 = &ldsA[(((wv    ) << 4) | l15) * LDA];
    const unsigned short* arow1 = &ldsA[(((wv + 8) << 4) | l15) * LDA];

#pragma unroll
    for (int kc = 0; kc < 8; ++kc) {
        // A fragments: lane<16 -> K = kb+0..7 / kb+16..23; lane>=16 -> +8.
        const int off = (kc << 5) | (hsel << 3);
        const v16bf a0 = ldsA_frag(arow0 + off);
        const v16bf a1 = ldsA_frag(arow1 + off);
#pragma unroll
        for (int nb = 0; nb < 8; ++nb) {
            const v16us bu =
                *(const v16us*)&ldsB[(((((nb << 3) | kc) << 5) | lane) << 4)];
            const v16bf b = __builtin_bit_cast(v16bf, bu);
            acc0[nb] = __builtin_amdgcn_wmma_f32_16x16x32_bf16(
                false, a0, false, b, (short)0, acc0[nb], false, false);
            acc1[nb] = __builtin_amdgcn_wmma_f32_16x16x32_bf16(
                false, a1, false, b, (short)0, acc1[nb], false, false);
        }
    }

    // ---- Phase 3: +b1, ReLU, dot W2, half-wave butterfly reduce, store ----
    float w2v[8], b1v[8];
#pragma unroll
    for (int nb = 0; nb < 8; ++nb) {
        w2v[nb] = ldsW2[(nb << 4) | l15];
        b1v[nb] = ldsB1[(nb << 4) | l15];
    }
    const float b2v = b2[0];

#pragma unroll
    for (int p = 0; p < 2; ++p) {
#pragma unroll
        for (int j = 0; j < 8; ++j) {
            float s = 0.f;
#pragma unroll
            for (int nb = 0; nb < 8; ++nb) {
                const float h = (p ? acc1[nb][j] : acc0[nb][j]) + b1v[nb];
                s = fmaf(fmaxf(h, 0.f), w2v[nb], s);
            }
            // lanes 0..15 hold edge (mblk*16+j), lanes 16..31 hold (mblk*16+8+j)
            s += __shfl_xor(s, 1);
            s += __shfl_xor(s, 2);
            s += __shfl_xor(s, 4);
            s += __shfl_xor(s, 8);
            const long long e = ebase + ((wv + 8 * p) << 4) + (hsel << 3) + j;
            if (l15 == 0 && e < (long long)E) out[e] = s + b2v;
        }
    }
}

extern "C" void kernel_launch(void* const* d_in, const int* in_sizes, int n_in,
                              void* d_out, int out_size, void* d_ws, size_t ws_size,
                              hipStream_t stream) {
    const float*     x_nc = (const float*)d_in[0];
    const float*     x_pr = (const float*)d_in[1];
    const long long* eidx = (const long long*)d_in[2];   // jnp.int64 per reference
    const float*     W1   = (const float*)d_in[3];
    const float*     b1   = (const float*)d_in[4];
    const float*     W2   = (const float*)d_in[5];
    const float*     b2   = (const float*)d_in[6];
    float*           out  = (float*)d_out;
    const int E = out_size;

    (void)in_sizes; (void)n_in; (void)d_ws; (void)ws_size;

    hipFuncSetAttribute((const void*)edge_mlp_wmma_kernel,
                        hipFuncAttributeMaxDynamicSharedMemorySize, SMEM_BYTES);

    dim3 grid((unsigned)((E + TILE_E - 1) / TILE_E)), block(256);
    hipLaunchKernelGGL(edge_mlp_wmma_kernel, grid, block, SMEM_BYTES, stream,
                       x_nc, x_pr, eidx, W1, b1, W2, b2, out, E);
}
